// FastDrumSynthesizer_7215545057361
// MI455X (gfx1250) — matrix operand order repaired and checked by reading, not compiled, not run on hardware
//
#include <hip/hip_runtime.h>

#define T_LEN   262144
#define R_LEN   4000
#define NCHUNK  126      // ceil((R+16)/32) tap chunks of 32
#define KERPAD  4096     // zero-padded kernel, kerpad[x] = ker[x-15]
#define SIGWIN  5040     // reversed signal window per wave (4 tiles + R tail)
#define SIGPAD  5056     // padded to multiple of 16 elements (32B)

typedef __attribute__((ext_vector_type(16))) _Float16     v16h;
typedef __attribute__((ext_vector_type(8)))  float        v8f;
typedef __attribute__((ext_vector_type(8)))  unsigned int v8u;

struct B32Bytes { uint4 a; uint4 b; };   // 32-byte carrier for bit_cast

__device__ __forceinline__ unsigned short f32_to_f16(float f) {
  return __builtin_bit_cast(unsigned short, (_Float16)f);  // v_cvt_f16_f32 (RNE)
}

__global__ __launch_bounds__(128)
void fir_wmma_kernel(const float* __restrict__ sig,
                     const float* __restrict__ ker,
                     float* __restrict__ out) {
  __shared__ unsigned short lds_ker[KERPAD];
  __shared__ unsigned short lds_sig[4 * SIGPAD];

  const int tid  = threadIdx.x;
  const int lane = tid & 31;
  const int wave = tid >> 5;
  const int ch   = blockIdx.x >> 6;    // 64 channels
  const int ww   = blockIdx.x & 63;    // 64 WGs per channel
  const int t0   = (ww * 4 + wave) * 1024;   // 1024 outputs per wave

  // ---- stage zero-padded f16 kernel (source of Toeplitz A), WG-shared -----
  const float* kch = ker + ch * R_LEN;
  for (int x = tid; x < KERPAD; x += 128) {
    unsigned short v = 0;
    if (x >= 15 && x < 15 + R_LEN) v = f32_to_f16(kch[x - 15]);
    lds_ker[x] = v;
  }

  // ---- stage reversed f16 signal window for this wave's 4 tiles -----------
  // sigrev[x] = sig[t0 + 1023 - x], causally zero-padded
  const float* sch   = sig + (long)ch * T_LEN;
  const int    wbase = wave * SIGPAD;
  for (int x = lane; x < SIGWIN; x += 32) {
    int   t = t0 + 1023 - x;
    float v = (t >= 0) ? sch[t] : 0.0f;
    lds_sig[wbase + x] = f32_to_f16(v);
  }
  __syncthreads();

  const int M     = lane & 15;          // A row (lanes 0-15 / 16-31 share M)
  const int Khalf = (lane >> 4) * 8;    // A K-half per ISA 16-bit A layout
  const int jb    = (lane >> 4) * 8;    // B K-half per ISA 16-bit B layout

  v8f acc[4] = {};                      // 4 output tiles per wave

  for (int c = 0; c < NCHUNK; ++c) {
    // A fragment: A[i][j] = kerpad[c*32 + i + j]; element e of lane ->
    // K = Khalf + e + (e>=8 ? 8 : 0)  (ISA 16-bit 16x32 A layout)
    const int ab = c * 32 + M + Khalf;
    v8u au;
#pragma unroll
    for (int v = 0; v < 8; ++v) {
      int idx = ab + 2 * v + ((v >= 4) ? 8 : 0);
      au[v] = (unsigned)lds_ker[idx] | ((unsigned)lds_ker[idx + 1] << 16);
    }
    v16h af = __builtin_bit_cast(v16h, au);

    // B fragments: f16 K = jb+0..7 and jb+16..23 per lane, 16B-aligned (two b128 loads)
    const int sb = wbase + 1008 - M * 16 + c * 32 + jb;
#pragma unroll
    for (int k = 0; k < 4; ++k) {
      const uint4* bp = (const uint4*)&lds_sig[sb - k * 256];
      B32Bytes bs; bs.a = bp[0]; bs.b = bp[2];
      v16h bf = __builtin_bit_cast(v16h, bs);
      acc[k] = __builtin_amdgcn_wmma_f32_16x16x32_f16(
          false, af, false, bf, (short)0, acc[k], false, false);
    }
  }

  // ---- transpose each 256-sample tile via LDS, coalesced b128 stores ------
  float*    osta = (float*)&lds_sig[wbase];   // reuse window region
  float*    och  = out + (long)ch * T_LEN;
  const int iof  = (lane >> 4) * 8;           // C/D layout: M = p + 8*(lane>=16)
  for (int k = 0; k < 4; ++k) {
    asm volatile("s_wait_dscnt 0" ::: "memory");
#pragma unroll
    for (int p = 0; p < 8; ++p)
      osta[M * 16 + iof + p] = acc[k][p];     // dword index = n*16 + i
    asm volatile("s_wait_dscnt 0" ::: "memory");
    float4 o0 = *(const float4*)&osta[lane * 8];
    float4 o1 = *(const float4*)&osta[lane * 8 + 4];
    int tb = t0 + k * 256 + lane * 8;
    *(float4*)&och[tb]     = o0;
    *(float4*)&och[tb + 4] = o1;
  }
}

extern "C" void kernel_launch(void* const* d_in, const int* in_sizes, int n_in,
                              void* d_out, int out_size, void* d_ws, size_t ws_size,
                              hipStream_t stream) {
  (void)in_sizes; (void)n_in; (void)d_ws; (void)ws_size; (void)out_size;
  const float* onsets  = (const float*)d_in[0];   // (8,8,262144) f32
  const float* sources = (const float*)d_in[1];   // (8,8,4000)   f32
  // d_in[2] = delay_compensation, unused by the reference as well
  float* out = (float*)d_out;                     // (8,8,262144) f32

  dim3 grid(64 * 64);   // 64 channels x 64 workgroups
  dim3 block(128);      // 4 wave32 per WG
  hipLaunchKernelGGL(fir_wmma_kernel, grid, block, 0, stream,
                     onsets, sources, out);
}
